// Concare_58239756533971
// MI455X (gfx1250) — matrix operand order repaired and checked by reading, not compiled, not run on hardware
//
#include <hip/hip_runtime.h>
#include <cstdint>
#include <cstddef>

typedef __attribute__((ext_vector_type(16))) _Float16 v16h;
typedef __attribute__((ext_vector_type(8)))  _Float16 v8h;
typedef __attribute__((ext_vector_type(8)))  float    v8f;
typedef int v4i __attribute__((vector_size(4 * sizeof(int))));

#define TT   128
#define BB   256
#define FFE  76
#define HHD  64
#define MR   19456    // B*F rows for the (·,64) activation matrices

#if defined(__HIP_DEVICE_COMPILE__) && \
    __has_builtin(__builtin_amdgcn_global_load_async_to_lds_b128) && \
    __has_builtin(__builtin_amdgcn_s_wait_asynccnt)
#define USE_ASYNC_LDS 1
#define GPTR(p) ((__attribute__((address_space(1))) v4i*)(p))
#define LPTR(p) ((__attribute__((address_space(3))) v4i*)(p))
#endif

// branch-free transcendentals: v_rcp_f32 + v_exp_f32, no IEEE-div sequences
__device__ __forceinline__ float rcpf(float x)  { return __builtin_amdgcn_rcpf(x); }
__device__ __forceinline__ float sigf(float x)  { return rcpf(1.0f + __expf(-x)); }
__device__ __forceinline__ float tanhfast(float x) { return 1.0f - 2.0f * rcpf(1.0f + __expf(2.0f * x)); }

__device__ __forceinline__ v16h cat8(v8h lo, v8h hi) {
  return __builtin_shufflevector(lo, hi, 0,1,2,3,4,5,6,7,8,9,10,11,12,13,14,15);
}

// ---------------------------------------------------------------------------
// K1: fused per-feature GRU scan. grid = F*2 (feature x batch-half), 256 thr.
// Dynamic LDS layout (bytes):
//   hF   [128*64] f32   @ 0        (32768)
//   hB0  [128*64] f16   @ 32768    (16384)
//   hB1  [128*64] f16   @ 49152    (16384)
//   Wl   [192*64] f16   @ 65536    (24576)
//   xl   [2*128]  f32   @ 90112    (1024)   -> total 91136
// ---------------------------------------------------------------------------
__global__ __launch_bounds__(256) void gru_scan(
    const float* __restrict__ x,      // (T,B,F)
    const float* __restrict__ w_ih,   // (F,3H)
    const float* __restrict__ w_hh,   // (F,3H,H)
    const float* __restrict__ b_ih,   // (F,3H)
    const float* __restrict__ b_hh,   // (F,3H)
    _Float16*    __restrict__ hs)     // (F,B,T,H) f16
{
  extern __shared__ __align__(16) char smem[];
  float*    hF  = (float*)smem;
  _Float16* hB0 = (_Float16*)(smem + 32768);
  _Float16* hB1 = (_Float16*)(smem + 49152);
  _Float16* Wl  = (_Float16*)(smem + 65536);
  float*    xl  = (float*)(smem + 90112);

  const int f    = blockIdx.x >> 1;
  const int half = blockIdx.x & 1;
  const int tid  = threadIdx.x;
  const int w    = tid >> 5, lane = tid & 31;
  const int j    = w & 3, msub = w >> 2;      // hidden-tile column, M-tile group
  const int ln   = lane & 15, kq = lane >> 4;
  const int nh   = j * 16 + ln;               // hidden index for this lane's N column

  // init hidden state + stage recurrent weights (converted f32->f16 once)
  for (int i = tid; i < 128 * 64; i += 256) { hF[i] = 0.f; hB0[i] = (_Float16)0.f; }
  for (int i = tid; i < 192 * 64; i += 256) Wl[i] = (_Float16)w_hh[(size_t)f * 12288 + i];

  // per-lane gate constants (column nh fixed for whole kernel)
  const float wr  = w_ih[f * 192 + nh];
  const float wz  = w_ih[f * 192 + 64 + nh];
  const float wn  = w_ih[f * 192 + 128 + nh];
  const float cr  = b_ih[f * 192 + nh]      + b_hh[f * 192 + nh];
  const float cz  = b_ih[f * 192 + 64 + nh] + b_hh[f * 192 + 64 + nh];
  const float bin = b_ih[f * 192 + 128 + nh];
  const float bhn = b_hh[f * 192 + 128 + nh];
  __syncthreads();

  // B fragments for W_hh, resident across the whole T scan.
  // B(k,n) = W_hh[g = N0 + lane%16][h = k]; lane holds 16 contiguous h values.
  v16h bf[3][2];
#pragma unroll
  for (int g = 0; g < 3; ++g)
#pragma unroll
    for (int ks = 0; ks < 2; ++ks)
      bf[g][ks] = *(const v16h*)(Wl + (g * 64 + nh) * 64 + ks * 32 + kq * 16);

  const int bbase = half * 128;

  for (int t = 0; t < TT; ++t) {
    const int cur = t & 1;
    _Float16* hBc = cur ? hB1 : hB0;   // read buffer (state at t)
    _Float16* hBn = cur ? hB0 : hB1;   // write buffer (state at t+1)
    if (tid < 128) xl[cur * 128 + tid] = x[((size_t)t * BB + bbase + tid) * FFE + f];
    __syncthreads();

#pragma unroll
    for (int m = 0; m < 4; ++m) {
      const int M0 = (msub * 4 + m) * 16;   // local batch-row tile base
      const _Float16* ap = hBc + (M0 + ln) * 64;
      v16h a0 = cat8(*(const v8h*)(ap + kq * 16),        *(const v8h*)(ap + kq * 16 + 8));
      v16h a1 = cat8(*(const v8h*)(ap + 32 + kq * 16),   *(const v8h*)(ap + 32 + kq * 16 + 8));
      v8f ar = {}, az = {}, an = {};
      ar = __builtin_amdgcn_wmma_f32_16x16x32_f16(false, a0, false, bf[0][0], (short)0, ar, false, false);
      ar = __builtin_amdgcn_wmma_f32_16x16x32_f16(false, a1, false, bf[0][1], (short)0, ar, false, false);
      az = __builtin_amdgcn_wmma_f32_16x16x32_f16(false, a0, false, bf[1][0], (short)0, az, false, false);
      az = __builtin_amdgcn_wmma_f32_16x16x32_f16(false, a1, false, bf[1][1], (short)0, az, false, false);
      an = __builtin_amdgcn_wmma_f32_16x16x32_f16(false, a0, false, bf[2][0], (short)0, an, false, false);
      an = __builtin_amdgcn_wmma_f32_16x16x32_f16(false, a1, false, bf[2][1], (short)0, an, false, false);

#pragma unroll
      for (int r = 0; r < 8; ++r) {
        const int row = M0 + r + kq * 8;   // local batch row (C/D layout)
        const float xv = xl[cur * 128 + row];
        const float rg = sigf(ar[r] + xv * wr + cr);
        const float zg = sigf(az[r] + xv * wz + cz);
        const float ng = tanhfast(xv * wn + bin + rg * (an[r] + bhn));
        const float hn = (1.f - zg) * ng + zg * hF[row * 64 + nh];
        hF[row * 64 + nh]  = hn;
        hBn[row * 64 + nh] = (_Float16)hn;
        hs[(((size_t)f * BB + bbase + row) * TT + t) * HHD + nh] = (_Float16)hn;
      }
    }
    // next iteration's barrier separates this step's writes from next reads
  }
}

// ---------------------------------------------------------------------------
// K2: per-feature time-aware SingleAttention. grid = F*32, 256 thr (wave=(f,b)).
// ---------------------------------------------------------------------------
__global__ __launch_bounds__(256) void feat_attn(
    const _Float16* __restrict__ hs,    // (F,B,T,H)
    const float* __restrict__ Wt,       // (F,H,8)
    const float* __restrict__ Wx,       // (F,H,8)
    const float* __restrict__ rate,     // (F,)
    float*    __restrict__ embF,        // (B,F,H) f32
    _Float16* __restrict__ embH)        // (B,F,H) f16
{
  __shared__ float WtL[512], WxL[512];
  __shared__ float uL[8][64];
  __shared__ float aL[8][128];
  const int f    = blockIdx.x >> 5;
  const int w    = threadIdx.x >> 5, lane = threadIdx.x & 31;
  const int b    = ((blockIdx.x & 31) << 3) + w;
  for (int i = threadIdx.x; i < 512; i += 256) { WtL[i] = Wt[f * 512 + i]; WxL[i] = Wx[f * 512 + i]; }
  __syncthreads();
  const float rs = sigf(rate[f]);
  const _Float16* hsb = hs + ((size_t)f * BB + b) * (TT * HHD);

  // q = h_last @ Wt   (8,)  -> u = Wx @ q  (64,)
  const float h0 = (float)hsb[127 * 64 + lane];
  const float h1 = (float)hsb[127 * 64 + 32 + lane];
  float q[8];
#pragma unroll
  for (int a = 0; a < 8; ++a) q[a] = h0 * WtL[lane * 8 + a] + h1 * WtL[(lane + 32) * 8 + a];
#pragma unroll
  for (int off = 16; off >= 1; off >>= 1)
#pragma unroll
    for (int a = 0; a < 8; ++a) q[a] += __shfl_xor(q[a], off);
  float u0 = 0.f, u1 = 0.f;
#pragma unroll
  for (int a = 0; a < 8; ++a) { u0 += WxL[lane * 8 + a] * q[a]; u1 += WxL[(lane + 32) * 8 + a] * q[a]; }
  uL[w][lane] = u0; uL[w][lane + 32] = u1;
  __syncthreads();
  float uu[64];
#pragma unroll
  for (int h = 0; h < 64; ++h) uu[h] = uL[w][h];

  // pass A: scores over T (lane handles 4 timesteps), wave softmax
  float e[4]; float mx = -1e30f;
#pragma unroll
  for (int it = 0; it < 4; ++it) {
    const int t = it * 32 + lane;
    float dp = 0.f;
#pragma unroll
    for (int c = 0; c < 8; ++c) {
      v8h hv = *(const v8h*)(hsb + t * 64 + c * 8);
#pragma unroll
      for (int ee = 0; ee < 8; ++ee) dp += (float)hv[ee] * uu[c * 8 + ee];
    }
    const float sg  = sigf(dp);
    const float den = rs * __logf(2.72f + (1.f - sg)) * (float)(TT - t);
    e[it] = fmaxf(sg * rcpf(den), 0.f);
    mx = fmaxf(mx, e[it]);
  }
#pragma unroll
  for (int off = 16; off >= 1; off >>= 1) mx = fmaxf(mx, __shfl_xor(mx, off));
  float sum = 0.f;
#pragma unroll
  for (int it = 0; it < 4; ++it) { e[it] = __expf(e[it] - mx); sum += e[it]; }
#pragma unroll
  for (int off = 16; off >= 1; off >>= 1) sum += __shfl_xor(sum, off);
  const float inv = rcpf(sum);
#pragma unroll
  for (int it = 0; it < 4; ++it) aL[w][it * 32 + lane] = e[it] * inv;
  __syncthreads();

  // pass B: emb[h] = sum_t a_t * hs[t][h]   (lane owns h, h+32)
  float a0 = 0.f, a1 = 0.f;
  for (int t = 0; t < TT; ++t) {
    if ((t & 7) == 0) __builtin_prefetch(hsb + (t + 16) * 64, 0, 0);  // global_prefetch_b8
    const float av = aL[w][t];
    a0 += av * (float)hsb[t * 64 + lane];
    a1 += av * (float)hsb[t * 64 + 32 + lane];
  }
  const size_t o = ((size_t)b * FFE + f) * HHD;
  embF[o + lane] = a0;               embF[o + 32 + lane] = a1;
  embH[o + lane] = (_Float16)a0;     embH[o + 32 + lane] = (_Float16)a1;
}

// ---------------------------------------------------------------------------
// Generic WMMA GEMM:  out(M,N) = act( A(M,K)f16 @ WtT(N,K)f16 + bias [+ resid] )
// grid = M/128 blocks, 256 thr; W^T staged in dynamic LDS (N*K*2 bytes) via
// GLOBAL_LOAD_ASYNC_TO_LDS_B128 when the toolchain exposes it.
// ---------------------------------------------------------------------------
__global__ __launch_bounds__(256) void gemm_f16(
    const _Float16* __restrict__ A,
    const _Float16* __restrict__ WtT,     // (N,K) f16, pre-transposed
    const float*    __restrict__ bias,    // (N,)
    const float*    __restrict__ resid,   // (M,N) or null
    float*          __restrict__ outF,    // (M,N) or null
    _Float16*       __restrict__ outH,    // (M,N) or null
    int N, int K, int relu)
{
  extern __shared__ __align__(16) _Float16 Wl[];
  const int tid = threadIdx.x, w = tid >> 5, lane = tid & 31;
  const int ln = lane & 15, kq = lane >> 4;
#if defined(USE_ASYNC_LDS)
  for (int c = tid; c < (N * K) >> 3; c += 256)     // 16-byte chunks
    __builtin_amdgcn_global_load_async_to_lds_b128(GPTR(WtT + c * 8), LPTR(Wl + c * 8), 0, 0);
  __builtin_amdgcn_s_wait_asynccnt(0);
#else
  for (int i = tid; i < N * K; i += 256) Wl[i] = WtT[i];
#endif
  __syncthreads();
  const int Mt = blockIdx.x * 8 + w;
  const _Float16* ap = A + (size_t)(Mt * 16 + ln) * K;

  for (int nt = 0; nt < (N >> 4); ++nt) {
    v8f acc = {};
    const _Float16* bp = Wl + (nt * 16 + ln) * K + kq * 16;
    for (int ks = 0; ks < (K >> 5); ++ks) {
      v16h a = cat8(*(const v8h*)(ap + ks * 32 + kq * 16),
                    *(const v8h*)(ap + ks * 32 + kq * 16 + 8));
      v16h b = *(const v16h*)(bp + ks * 32);
      acc = __builtin_amdgcn_wmma_f32_16x16x32_f16(false, a, false, b, (short)0, acc, false, false);
    }
    const int col = nt * 16 + ln;
#pragma unroll
    for (int r = 0; r < 8; ++r) {
      const int row = Mt * 16 + r + kq * 8;
      float v = acc[r] + bias[col];
      if (resid) v += resid[(size_t)row * N + col];
      if (relu)  v = fmaxf(v, 0.f);
      if (outF)  outF[(size_t)row * N + col] = v;
      if (outH)  outH[(size_t)row * N + col] = (_Float16)v;
    }
  }
}

// ---------------------------------------------------------------------------
// K4: per-(batch,head) self-attention over the F axis (76x16). grid = B*4.
// ---------------------------------------------------------------------------
__global__ __launch_bounds__(128) void mha_core(
    const float* __restrict__ qP, const float* __restrict__ kP,
    const float* __restrict__ vP, _Float16* __restrict__ ctxH)
{
  __shared__ __align__(16) float kT[76 * 16];
  __shared__ __align__(16) float vT[76 * 16];
  const int b = blockIdx.x >> 2, hh = blockIdx.x & 3;
  const int tid = threadIdx.x;
#if defined(USE_ASYNC_LDS)
  for (int c = tid; c < 304; c += 128) {            // 16-byte chunks of 76x16 f32
    const int row = c >> 2, q4 = c & 3;
    const size_t src = ((size_t)b * FFE + row) * HHD + hh * 16 + q4 * 4;
    __builtin_amdgcn_global_load_async_to_lds_b128(GPTR(kP + src), LPTR(kT + c * 4), 0, 0);
    __builtin_amdgcn_global_load_async_to_lds_b128(GPTR(vP + src), LPTR(vT + c * 4), 0, 0);
  }
  __builtin_amdgcn_s_wait_asynccnt(0);
#else
  for (int i = tid; i < 76 * 16; i += 128) {
    const size_t src = ((size_t)b * FFE + (i >> 4)) * HHD + hh * 16 + (i & 15);
    kT[i] = kP[src]; vT[i] = vP[src];
  }
#endif
  __syncthreads();
  if (tid < 76) {
    float qr[16];
#pragma unroll
    for (int jj = 0; jj < 16; ++jj) qr[jj] = qP[((size_t)b * FFE + tid) * HHD + hh * 16 + jj];
    float sc[76]; float mx = -1e30f;
#pragma unroll
    for (int c = 0; c < 76; ++c) {
      float d = 0.f;
#pragma unroll
      for (int jj = 0; jj < 16; ++jj) d += qr[jj] * kT[c * 16 + jj];
      sc[c] = d * 0.25f;                 // 1/sqrt(dk=16)
      mx = fmaxf(mx, sc[c]);
    }
    float s = 0.f;
#pragma unroll
    for (int c = 0; c < 76; ++c) { sc[c] = __expf(sc[c] - mx); s += sc[c]; }
    const float inv = rcpf(s);
    float cx[16];
#pragma unroll
    for (int jj = 0; jj < 16; ++jj) cx[jj] = 0.f;
#pragma unroll
    for (int c = 0; c < 76; ++c) {
      const float av = sc[c] * inv;
#pragma unroll
      for (int jj = 0; jj < 16; ++jj) cx[jj] += av * vT[c * 16 + jj];
    }
#pragma unroll
    for (int jj = 0; jj < 16; ++jj)
      ctxH[((size_t)b * FFE + tid) * HHD + hh * 16 + jj] = (_Float16)cx[jj];
  }
}

// ---------------------------------------------------------------------------
// K5: FinalAttentionQKV('mul') + output head. grid = 32, wave = one batch row.
// ---------------------------------------------------------------------------
__global__ __launch_bounds__(256) void final_head(
    const float* __restrict__ ctx2F, const float* __restrict__ fkF, const float* __restrict__ fvF,
    const float* __restrict__ fa_wq, const float* __restrict__ fa_bq,
    const float* __restrict__ w0, const float* __restrict__ b0,
    const float* __restrict__ w1, const float* __restrict__ b1,
    float* __restrict__ out)
{
  __shared__ float fqL[8][64], awL[8][80], vL[8][64];
  const int w = threadIdx.x >> 5, lane = threadIdx.x & 31;
  const int b = blockIdx.x * 8 + w;

  const float* last = ctx2F + ((size_t)b * FFE + 75) * HHD;
  float f0 = fa_bq[lane], f1 = fa_bq[lane + 32];
  for (int k = 0; k < 64; ++k) {
    const float cv = last[k];
    f0 += cv * fa_wq[k * 64 + lane];
    f1 += cv * fa_wq[k * 64 + lane + 32];
  }
  fqL[w][lane] = f0; fqL[w][lane + 32] = f1;
  __syncthreads();

  float fe[3]; float mx = -1e30f;
#pragma unroll
  for (int i = 0; i < 3; ++i) {
    const int f = i * 32 + lane;
    if (f < 76) {
      float d = 0.f;
      for (int jj = 0; jj < 64; ++jj) d += fkF[((size_t)b * FFE + f) * HHD + jj] * fqL[w][jj];
      fe[i] = d; mx = fmaxf(mx, d);
    } else fe[i] = -1e30f;
  }
#pragma unroll
  for (int off = 16; off >= 1; off >>= 1) mx = fmaxf(mx, __shfl_xor(mx, off));
  float s = 0.f;
#pragma unroll
  for (int i = 0; i < 3; ++i) {
    const int f = i * 32 + lane;
    if (f < 76) { fe[i] = __expf(fe[i] - mx); s += fe[i]; } else fe[i] = 0.f;
  }
#pragma unroll
  for (int off = 16; off >= 1; off >>= 1) s += __shfl_xor(s, off);
  const float inv = rcpf(s);
#pragma unroll
  for (int i = 0; i < 3; ++i) { const int f = i * 32 + lane; if (f < 76) awL[w][f] = fe[i] * inv; }
  __syncthreads();

  float v0 = 0.f, v1 = 0.f;
  for (int f = 0; f < 76; ++f) {
    const float av = awL[w][f];
    v0 += av * fvF[((size_t)b * FFE + f) * HHD + lane];
    v1 += av * fvF[((size_t)b * FFE + f) * HHD + lane + 32];
  }
  vL[w][lane] = v0; vL[w][lane + 32] = v1;
  __syncthreads();

  float o0 = b0[lane], o1 = b0[lane + 32];
  for (int jj = 0; jj < 64; ++jj) {
    const float vv = vL[w][jj];
    o0 += vv * w0[jj * 64 + lane];
    o1 += vv * w0[jj * 64 + lane + 32];
  }
  o0 = fmaxf(o0, 0.f); o1 = fmaxf(o1, 0.f);
  float p = o0 * w1[lane] + o1 * w1[lane + 32];
#pragma unroll
  for (int off = 16; off >= 1; off >>= 1) p += __shfl_xor(p, off);
  if (lane == 0) out[b] = sigf(p + b1[0]);
}

// (K,N) f32 -> (N,K) f16 weight transpose for WMMA B-operand staging
__global__ void transp_h(const float* __restrict__ s, _Float16* __restrict__ d, int K, int N) {
  const int idx = blockIdx.x * 256 + threadIdx.x;
  if (idx < K * N) { const int k = idx / N, n = idx - k * N; d[n * K + k] = (_Float16)s[idx]; }
}

// ---------------------------------------------------------------------------
extern "C" void kernel_launch(void* const* d_in, const int* in_sizes, int n_in,
                              void* d_out, int out_size, void* d_ws, size_t ws_size,
                              hipStream_t stream) {
  (void)in_sizes; (void)n_in; (void)out_size; (void)ws_size;
  const float* x        = (const float*)d_in[0];
  const float* gru_w_ih = (const float*)d_in[1];
  const float* gru_w_hh = (const float*)d_in[2];
  const float* gru_b_ih = (const float*)d_in[3];
  const float* gru_b_hh = (const float*)d_in[4];
  const float* att_Wt   = (const float*)d_in[5];
  const float* att_Wx   = (const float*)d_in[6];
  const float* att_rate = (const float*)d_in[7];
  const float* mha_wq = (const float*)d_in[8];  const float* mha_bq = (const float*)d_in[9];
  const float* mha_wk = (const float*)d_in[10]; const float* mha_bk = (const float*)d_in[11];
  const float* mha_wv = (const float*)d_in[12]; const float* mha_bv = (const float*)d_in[13];
  const float* mha_wo = (const float*)d_in[14]; const float* mha_bo = (const float*)d_in[15];
  const float* ffn_w1 = (const float*)d_in[16]; const float* ffn_b1 = (const float*)d_in[17];
  const float* ffn_w2 = (const float*)d_in[18]; const float* ffn_b2 = (const float*)d_in[19];
  const float* fa_wq  = (const float*)d_in[20]; const float* fa_bq  = (const float*)d_in[21];
  const float* fa_wk  = (const float*)d_in[22]; const float* fa_bk  = (const float*)d_in[23];
  const float* fa_wv  = (const float*)d_in[24]; const float* fa_bv  = (const float*)d_in[25];
  const float* out0_w = (const float*)d_in[26]; const float* out0_b = (const float*)d_in[27];
  const float* out1_w = (const float*)d_in[28]; const float* out1_b = (const float*)d_in[29];

  char* ws = (char*)d_ws;
  size_t cur = 0;
  auto alloc = [&](size_t bytes) -> void* {
    void* p = (void*)(ws + cur);
    cur = (cur + bytes + 255) & ~((size_t)255);
    return p;
  };
  _Float16* hsH   = (_Float16*)alloc((size_t)FFE * BB * TT * HHD * 2);  // 319 MB
  float*    embF  = (float*)alloc((size_t)MR * 64 * 4);
  _Float16* embH  = (_Float16*)alloc((size_t)MR * 64 * 2);
  float*    qP    = (float*)alloc((size_t)MR * 64 * 4);
  float*    kP    = (float*)alloc((size_t)MR * 64 * 4);
  float*    vP    = (float*)alloc((size_t)MR * 64 * 4);
  _Float16* ctxAH = (_Float16*)alloc((size_t)MR * 64 * 2);
  float*    ctxF  = (float*)alloc((size_t)MR * 64 * 4);
  _Float16* ctxH  = (_Float16*)alloc((size_t)MR * 64 * 2);
  _Float16* ffn1H = (_Float16*)alloc((size_t)MR * 256 * 2);
  float*    ctx2F = (float*)alloc((size_t)MR * 64 * 4);
  _Float16* ctx2H = (_Float16*)alloc((size_t)MR * 64 * 2);
  float*    fkF   = (float*)alloc((size_t)MR * 64 * 4);
  float*    fvF   = (float*)alloc((size_t)MR * 64 * 4);
  _Float16* wqT   = (_Float16*)alloc(64 * 64 * 2);
  _Float16* wkT   = (_Float16*)alloc(64 * 64 * 2);
  _Float16* wvT   = (_Float16*)alloc(64 * 64 * 2);
  _Float16* woT   = (_Float16*)alloc(64 * 64 * 2);
  _Float16* w1T   = (_Float16*)alloc(64 * 256 * 2);
  _Float16* w2T   = (_Float16*)alloc(256 * 64 * 2);
  _Float16* fkT   = (_Float16*)alloc(64 * 64 * 2);
  _Float16* fvT   = (_Float16*)alloc(64 * 64 * 2);

  // weight transposes -> f16 (N,K)
  transp_h<<<16, 256, 0, stream>>>(mha_wq, wqT, 64, 64);
  transp_h<<<16, 256, 0, stream>>>(mha_wk, wkT, 64, 64);
  transp_h<<<16, 256, 0, stream>>>(mha_wv, wvT, 64, 64);
  transp_h<<<16, 256, 0, stream>>>(mha_wo, woT, 64, 64);
  transp_h<<<64, 256, 0, stream>>>(ffn_w1, w1T, 64, 256);
  transp_h<<<64, 256, 0, stream>>>(ffn_w2, w2T, 256, 64);
  transp_h<<<16, 256, 0, stream>>>(fa_wk, fkT, 64, 64);
  transp_h<<<16, 256, 0, stream>>>(fa_wv, fvT, 64, 64);

  // 1) GRU scan (WMMA)         2) per-feature attention
  gru_scan<<<FFE * 2, 256, 91136, stream>>>(x, gru_w_ih, gru_w_hh, gru_b_ih, gru_b_hh, hsH);
  feat_attn<<<FFE * 32, 256, 0, stream>>>(hsH, att_Wt, att_Wx, att_rate, embF, embH);

  // 3) MHA over features + residual
  gemm_f16<<<MR / 128, 256, 64 * 64 * 2, stream>>>(embH, wqT, mha_bq, nullptr, qP, nullptr, 64, 64, 0);
  gemm_f16<<<MR / 128, 256, 64 * 64 * 2, stream>>>(embH, wkT, mha_bk, nullptr, kP, nullptr, 64, 64, 0);
  gemm_f16<<<MR / 128, 256, 64 * 64 * 2, stream>>>(embH, wvT, mha_bv, nullptr, vP, nullptr, 64, 64, 0);
  mha_core<<<BB * 4, 128, 0, stream>>>(qP, kP, vP, ctxAH);
  gemm_f16<<<MR / 128, 256, 64 * 64 * 2, stream>>>(ctxAH, woT, mha_bo, embF, ctxF, ctxH, 64, 64, 0);

  // 4) FFN + residual
  gemm_f16<<<MR / 128, 256, 64 * 256 * 2, stream>>>(ctxH, w1T, ffn_b1, nullptr, nullptr, ffn1H, 256, 64, 1);
  gemm_f16<<<MR / 128, 256, 256 * 64 * 2, stream>>>(ffn1H, w2T, ffn_b2, ctxF, ctx2F, ctx2H, 64, 256, 0);

  // 5) final attention projections + head
  gemm_f16<<<MR / 128, 256, 64 * 64 * 2, stream>>>(ctx2H, fkT, fa_bk, nullptr, fkF, nullptr, 64, 64, 0);
  gemm_f16<<<MR / 128, 256, 64 * 64 * 2, stream>>>(ctx2H, fvT, fa_bv, nullptr, fvF, nullptr, 64, 64, 0);
  final_head<<<32, 256, 0, stream>>>(ctx2F, fkF, fvF, fa_wq, fa_bq, out0_w, out0_b, out1_w, out1_b,
                                     (float*)d_out);
}
